// ContinuousSparsemax_15066745274926
// MI455X (gfx1250) — matrix operand order repaired and checked by reading, compiled, not run on hardware
//
#include <hip/hip_runtime.h>
#include <math.h>

typedef __attribute__((ext_vector_type(16))) _Float16 v16h;
typedef __attribute__((ext_vector_type(8)))  float    v8f;

#define BDIM   4096
#define MDIM   128      // alpha cols (GEMM1 K)
#define NBAS   512
#define DX     100
#define THS    130      // theta row stride (M+2)
#define TPAD   112      // GEMM1 N padded: 7 tiles of 16
#define NT1    7        // GEMM1 N tiles
#define KSTEPS 4        // both GEMMs: K=128 = 4 x 32

#define TSTEP  (10.0f / 99.0f)   // linspace(-5,5,100) step

__device__ __forceinline__ float Tval(int t) {
  return fmaf((float)t, TSTEP, -5.0f);     // no division -> no v_div_* chain
}
__device__ __forceinline__ float wval(int t) {
  return (t == 0 || t == 99) ? 0.5f * TSTEP : TSTEP;
}

// ---------------------------------------------------------------------------
// Prep: write K (128x112) and phi1^T (128x512) as f16 WMMA B-fragments.
// B-frag layout (32x16, 16-bit, wave32): element (lane,j) <-> k = 16*(j/8) + 8*(lane/16) + (j%8),
// col = lane%16. One fragment = 512 halves, contiguous: frag[lane*16 + j].
// ---------------------------------------------------------------------------
__global__ __launch_bounds__(256) void csmax_prep_kernel(
    const float* __restrict__ mu, const float* __restrict__ sigma,
    _Float16* __restrict__ kfrag, _Float16* __restrict__ pfrag) {
  int idx = blockIdx.x * 256 + threadIdx.x;
  const int NK = NT1 * KSTEPS * 512;           // 14336
  if (idx < NK) {
    int j = idx & 15, lane = (idx >> 4) & 31, frag = idx >> 9;
    int ks = frag & 3, ntile = frag >> 2;
    int t = ntile * 16 + (lane & 15);
    int k = ks * 32 + ((j >> 3) << 4) + ((lane >> 4) << 3) + (j & 7);  // k < 128 always
    float v = 0.0f;
    if (t < DX) {
      float loc = (float)k / 127.0f;
      float d = loc - Tval(t);
      v = __expf(-50.0f * d * d);              // exp(-0.5 d^2 / 0.01)
    }
    kfrag[idx] = (_Float16)v;
  }
  int idx2 = idx - NK;
  const int NP = 32 * KSTEPS * 512;            // 65536
  if (idx2 >= 0 && idx2 < NP) {
    int j = idx2 & 15, lane = (idx2 >> 4) & 31, frag = idx2 >> 9;
    int ks = frag & 3, ntile = frag >> 2;
    int n = ntile * 16 + (lane & 15);          // basis index
    int t = ks * 32 + ((j >> 3) << 4) + ((lane >> 4) << 3) + (j & 7);  // padded time index
    float v = 0.0f;
    if (t < DX) {
      float s = sigma[n];
      float z = (mu[n] - Tval(t)) / s;
      v = 0.3989422804014327f * __expf(-0.5f * z * z) / s;
    }
    pfrag[idx2] = (_Float16)v;
  }
}

// ---------------------------------------------------------------------------
// Fused main kernel: one 16-row strip per block, 8 waves (256 threads).
//   GEMM1 (alpha@K) -> elementwise -> trapz normalize -> GEMM2 (density@phi1^T)
// A-frag layout (16x32, 16-bit, wave32): (lane,j) <-> row = lane%16,
// k = 16*(j/8) + 8*(lane/16) + (j%8).
// expbuf stores ew = max(1 + f - q, 1e-8) * w[t]  (weight pre-folded).
// ---------------------------------------------------------------------------
__global__ __launch_bounds__(256) void csmax_main_kernel(
    const float* __restrict__ theta,
    const _Float16* __restrict__ kfrag,
    const _Float16* __restrict__ pfrag,
    float* __restrict__ out) {
  __shared__ alignas(32) _Float16 aFrag[4 * 512];   // 4 A-fragments (4 KB)
  __shared__ float expbuf[16 * TPAD];
  __shared__ float zpart[16 * 16];
  __shared__ float zinv_s[16];
  __shared__ float muc_s[16], nth1_s[16];           // nth1 = -theta[:,1] = 1/(2*ssq)

  const int tid   = threadIdx.x;
  const int lane  = tid & 31;
  const int wave  = tid >> 5;
  const int row0  = blockIdx.x * 16;

  // ---- stage alpha strip as f16 A-fragments in LDS ----
#pragma unroll
  for (int i = 0; i < 8; ++i) {
    int s  = tid + i * 256;                 // 2048 slots
    int j  = s & 15, ln = (s >> 4) & 31, fr = s >> 9;
    int m  = ln & 15;
    int k  = fr * 32 + ((j >> 3) << 4) + ((ln >> 4) << 3) + (j & 7);
    aFrag[s] = (_Float16)theta[(row0 + m) * THS + 2 + k];
  }
  if (tid < 16) {
    float th0 = theta[(row0 + tid) * THS + 0];
    float th1 = theta[(row0 + tid) * THS + 1];   // <= -0.5
    nth1_s[tid] = -th1;                          // q = -th1 * dmu^2
    muc_s[tid]  = th0 * (-0.5f / th1);           // one division per row
  }
  __syncthreads();

  v16h a0 = *(const v16h*)(aFrag + 0 * 512 + lane * 16);
  v16h a1 = *(const v16h*)(aFrag + 1 * 512 + lane * 16);
  v16h a2 = *(const v16h*)(aFrag + 2 * 512 + lane * 16);
  v16h a3 = *(const v16h*)(aFrag + 3 * 512 + lane * 16);

  // ---- GEMM1: f = alpha @ K  (waves 0..6 cover 7 column tiles) ----
  if (wave < NT1) {
    const _Float16* kb = kfrag + (wave * KSTEPS) * 512 + lane * 16;
    v16h b0 = *(const v16h*)(kb + 0 * 512);
    v16h b1 = *(const v16h*)(kb + 1 * 512);
    v16h b2 = *(const v16h*)(kb + 2 * 512);
    v16h b3 = *(const v16h*)(kb + 3 * 512);
    v8f c = {};
    c = __builtin_amdgcn_wmma_f32_16x16x32_f16(false, a0, false, b0, (short)0, c, false, false);
    c = __builtin_amdgcn_wmma_f32_16x16x32_f16(false, a1, false, b1, (short)0, c, false, false);
    c = __builtin_amdgcn_wmma_f32_16x16x32_f16(false, a2, false, b2, (short)0, c, false, false);
    c = __builtin_amdgcn_wmma_f32_16x16x32_f16(false, a3, false, b3, (short)0, c, false, false);

    int   t  = wave * 16 + (lane & 15);
    float Tt = Tval(t);
    float wv = (t < DX) ? wval(t) : 0.0f;   // folds weight AND pads t>=100 to 0
#pragma unroll
    for (int r = 0; r < 8; ++r) {
      int   m   = r + ((lane >> 4) << 3);
      float dmu = muc_s[m] - Tt;
      float q   = nth1_s[m] * dmu * dmu;
      float e   = fmaxf(1.0f + c[r] - q, 1e-8f);
      expbuf[m * TPAD + t] = e * wv;
    }
  }
  __syncthreads();

  // ---- parallel trapezoid normalizer: 16 rows x 16 partials of 7 ----
  {
    int m = tid & 15, part = tid >> 4;      // 256 threads -> 16x16
    float z = 0.0f;
#pragma unroll
    for (int i = 0; i < 7; ++i) {
      int t = part * 7 + i;                 // covers 0..111 (t>=100 slots are 0)
      z += expbuf[m * TPAD + t];
    }
    zpart[part * 16 + m] = z;
  }
  __syncthreads();
  if (tid < 16) {
    float z = 0.0f;
#pragma unroll
    for (int p = 0; p < 16; ++p) z += zpart[p * 16 + tid];
    zinv_s[tid] = 1.0f / z;
  }
  __syncthreads();

  // ---- rebuild aFrag as density (16x128 padded, f16 A-fragments) ----
#pragma unroll
  for (int i = 0; i < 8; ++i) {
    int s = tid + i * 256;
    int j = s & 15, ln = (s >> 4) & 31, fr = s >> 9;
    int m = ln & 15;
    int t = fr * 32 + ((j >> 3) << 4) + ((ln >> 4) << 3) + (j & 7);
    float v = 0.0f;
    if (t < DX) v = expbuf[m * TPAD + t] * zinv_s[m];   // ew/Z
    aFrag[s] = (_Float16)v;
  }
  __syncthreads();

  a0 = *(const v16h*)(aFrag + 0 * 512 + lane * 16);
  a1 = *(const v16h*)(aFrag + 1 * 512 + lane * 16);
  a2 = *(const v16h*)(aFrag + 2 * 512 + lane * 16);
  a3 = *(const v16h*)(aFrag + 3 * 512 + lane * 16);

  // ---- GEMM2: out = density @ phi1^T  (each wave: 4 of 32 N-tiles) ----
#pragma unroll
  for (int i = 0; i < 4; ++i) {
    int ntile = wave * 4 + i;
    const _Float16* pb = pfrag + (ntile * KSTEPS) * 512 + lane * 16;
    v16h b0 = *(const v16h*)(pb + 0 * 512);
    v16h b1 = *(const v16h*)(pb + 1 * 512);
    v16h b2 = *(const v16h*)(pb + 2 * 512);
    v16h b3 = *(const v16h*)(pb + 3 * 512);
    v8f acc = {};
    acc = __builtin_amdgcn_wmma_f32_16x16x32_f16(false, a0, false, b0, (short)0, acc, false, false);
    acc = __builtin_amdgcn_wmma_f32_16x16x32_f16(false, a1, false, b1, (short)0, acc, false, false);
    acc = __builtin_amdgcn_wmma_f32_16x16x32_f16(false, a2, false, b2, (short)0, acc, false, false);
    acc = __builtin_amdgcn_wmma_f32_16x16x32_f16(false, a3, false, b3, (short)0, acc, false, false);

    int n = ntile * 16 + (lane & 15);
#pragma unroll
    for (int r = 0; r < 8; ++r) {
      int m = r + ((lane >> 4) << 3);
      out[(row0 + m) * NBAS + n] = acc[r];
    }
  }
}

extern "C" void kernel_launch(void* const* d_in, const int* in_sizes, int n_in,
                              void* d_out, int out_size, void* d_ws, size_t ws_size,
                              hipStream_t stream) {
  (void)in_sizes; (void)n_in; (void)out_size; (void)ws_size;
  const float* theta = (const float*)d_in[0];
  const float* mu    = (const float*)d_in[1];
  const float* sigma = (const float*)d_in[2];
  float* out = (float*)d_out;

  _Float16* kfrag = (_Float16*)d_ws;                         // 28 KB
  _Float16* pfrag = (_Float16*)((char*)d_ws + 32768);        // 128 KB

  const int prep_slots = NT1 * KSTEPS * 512 + 32 * KSTEPS * 512;   // 79872
  csmax_prep_kernel<<<(prep_slots + 255) / 256, 256, 0, stream>>>(mu, sigma, kfrag, pfrag);
  csmax_main_kernel<<<BDIM / 16, 256, 0, stream>>>(theta, kfrag, pfrag, out);
}
